// GNNSelector_63247688401688
// MI455X (gfx1250) — matrix operand order repaired and checked by reading, not compiled, not run on hardware
//
#include <hip/hip_runtime.h>
#include <hip/hip_bf16.h>

#define N0 100000
#define N1 40000
#define N2 20000
#define DEG 16
#define DIN 256

typedef _Float16 half_t;
typedef __attribute__((ext_vector_type(16))) _Float16 v16h;
typedef __attribute__((ext_vector_type(8)))  _Float16 v8h;
typedef __attribute__((ext_vector_type(8)))  float    v8f;

// ---------------------------------------------------------------- utilities

static __device__ __forceinline__ float wave_sum32(float v) {
  #pragma unroll
  for (int off = 16; off > 0; off >>= 1) v += __shfl_xor(v, off, 32);
  return v;
}

__global__ void f32_to_f16_kernel(const float* __restrict__ src,
                                  half_t* __restrict__ dst, int n) {
  int i = blockIdx.x * blockDim.x + threadIdx.x;
  if (i < n) dst[i] = (half_t)src[i];
}

// W is [K=256][N=256] row-major fp32; produce Wt[n][k] f16 (column-major view)
__global__ void wtrans_kernel(const float* __restrict__ W,
                              half_t* __restrict__ Wt) {
  int i = blockIdx.x * blockDim.x + threadIdx.x;   // 65536 threads
  int n = i >> 8, k = i & 255;
  Wt[n * 256 + k] = (half_t)W[k * 256 + n];
}

// ---------------------------------------------------------------- GEMV (Wout)

__global__ void gemv256_kernel(const float* __restrict__ A,
                               const float* __restrict__ w,
                               const float* __restrict__ bias,
                               float* out, float* out2, int M, int M2) {
  int wid  = (blockIdx.x * blockDim.x + threadIdx.x) >> 5;
  int lane = threadIdx.x & 31;
  if (wid >= M) return;
  const float* a  = A + (size_t)wid * DIN + lane * 8;
  const float* ww = w + lane * 8;
  float p = 0.f;
  #pragma unroll
  for (int i = 0; i < 8; i++) p += a[i] * ww[i];
  p = wave_sum32(p);
  if (lane == 0) {
    float v = p + bias[0];
    if (out)  out[wid] = v;
    if (out2 && wid < M2) out2[wid] = v;
  }
}

// ---------------------------------------------------------------- pool (layer0)

__global__ void pool0_kernel(const float* __restrict__ l0,
                             const int* __restrict__ esrc,
                             int* __restrict__ newsrc) {
  int n = blockIdx.x * blockDim.x + threadIdx.x;
  if (n >= N1) return;
  float sd = 1.f / (1.f + __expf(-l0[n]));
  float sim[DEG]; int s[DEG]; bool taken[DEG];
  #pragma unroll
  for (int e = 0; e < DEG; e++) {
    int sc = esrc[n * DEG + e];
    s[e] = sc;
    float ss = 1.f / (1.f + __expf(-l0[sc]));
    sim[e] = 1.f - fabsf(ss - sd);
    taken[e] = false;
  }
  // stable top-8 (ties -> lowest index), matching jax.lax.top_k
  for (int j = 0; j < 8; j++) {
    int bi = 0; float bv = -1e30f;
    #pragma unroll
    for (int e = 0; e < DEG; e++)
      if (!taken[e] && sim[e] > bv) { bv = sim[e]; bi = e; }
    taken[bi] = true;
    newsrc[n * 8 + j] = s[bi];
  }
}

// ---------------------------------------------------------------- WMMA GEMM
// C[M,256] = A[M,256](f16) @ W[256,256] + bias, W given as Wt[n][k] f16.
// One wave computes a 32x64 tile (two 16-row A fragments share each B
// fragment -> half the B traffic, 8 WMMAs per B-load wait).
// M must be a multiple of 32 (100000/40000/20000 all are).

__global__ void __launch_bounds__(256)
gemm256_wmma_kernel(const half_t* __restrict__ A,
                    const half_t* __restrict__ Wt,
                    const float*  __restrict__ bias,
                    float* __restrict__ Cf, half_t* __restrict__ Ch, int M) {
  int wid  = (blockIdx.x * blockDim.x + threadIdx.x) >> 5;
  int lane = threadIdx.x & 31;
  int mtiles = M >> 5;                     // 32-row tiles
  int total  = mtiles * 4;                 // 4 column groups of 64
  if (wid >= total) return;                // wave-uniform: EXEC stays all-ones
  int mt = wid >> 2;
  int ng = wid & 3;
  int hi = (lane >= 16) ? 1 : 0;
  int mrow0 = mt * 32 + (lane & 15);
  int ncol0 = ng * 64 + (lane & 15);
  const half_t* arow0 = A + (size_t)mrow0 * DIN;
  const half_t* arow1 = arow0 + (size_t)16 * DIN;

  v8f acc0[4] = {};
  v8f acc1[4] = {};

  for (int kb = 0; kb < DIN; kb += 32) {
    // A fragments: 16x32, lanes<16 K={kb..kb+7, kb+16..kb+23}, lanes>=16 +8
    v16h af0, af1;
    {
      v8h g0 = *(const v8h*)(arow0 + kb + hi * 8);
      v8h g1 = *(const v8h*)(arow0 + kb + 16 + hi * 8);
      v8h h0 = *(const v8h*)(arow1 + kb + hi * 8);
      v8h h1 = *(const v8h*)(arow1 + kb + 16 + hi * 8);
      #pragma unroll
      for (int i = 0; i < 8; i++) {
        af0[i] = g0[i]; af0[i + 8] = g1[i];
        af1[i] = h0[i]; af1[i + 8] = h1[i];
      }
    }
    #pragma unroll
    for (int j = 0; j < 4; j++) {
      // B fragment: 32x16, lane col=lane&15, K={kb..kb+7, kb+16..kb+23}, lanes>=16 +8
      const half_t* bcol = Wt + (size_t)(ncol0 + j * 16) * DIN + kb + hi * 8;
      v16h bf;
      v8h b0 = *(const v8h*)(bcol);
      v8h b1 = *(const v8h*)(bcol + 16);
      #pragma unroll
      for (int i = 0; i < 8; i++) { bf[i] = b0[i]; bf[i + 8] = b1[i]; }
      acc0[j] = __builtin_amdgcn_wmma_f32_16x16x32_f16(
          false, af0, false, bf, (short)0, acc0[j], false, false);
      acc1[j] = __builtin_amdgcn_wmma_f32_16x16x32_f16(
          false, af1, false, bf, (short)0, acc1[j], false, false);
    }
  }

  #pragma unroll
  for (int j = 0; j < 4; j++) {
    int n  = ncol0 + j * 16;
    float bn = bias ? bias[n] : 0.f;
    #pragma unroll
    for (int i = 0; i < 8; i++) {
      int m0 = mt * 32 + i + hi * 8;       // first 16-row tile
      int m1 = m0 + 16;                    // second 16-row tile
      float v0 = acc0[j][i] + bn;
      float v1 = acc1[j][i] + bn;
      if (Cf) { Cf[(size_t)m0 * DIN + n] = v0; Cf[(size_t)m1 * DIN + n] = v1; }
      if (Ch) { Ch[(size_t)m0 * DIN + n] = (half_t)v0; Ch[(size_t)m1 * DIN + n] = (half_t)v1; }
    }
  }
}

// ---------------------------------------------------------------- attention
// One wave per dst node. Lane L owns dims [L*8, L*8+8): lanes 0-15 = head0,
// lanes 16-31 = head1. Online softmax over `deg` in-edges.

__global__ void attn_kernel(const float*  __restrict__ Q,
                            const half_t* __restrict__ K,
                            const half_t* __restrict__ V,
                            const float*  __restrict__ S,
                            const int*    __restrict__ srcs, int deg,
                            float* __restrict__ out, int nd) {
  int wid  = (blockIdx.x * blockDim.x + threadIdx.x) >> 5;
  int lane = threadIdx.x & 31;
  if (wid >= nd) return;
  const float scale = 0.08838834764831845f;   // 1/sqrt(128)

  float qr[8];
  const float* qp = Q + (size_t)wid * DIN + lane * 8;
  #pragma unroll
  for (int i = 0; i < 8; i++) qr[i] = qp[i];

  float mrun = -1e30f, ssum = 0.f;
  float acc[8] = {0.f, 0.f, 0.f, 0.f, 0.f, 0.f, 0.f, 0.f};

  for (int e = 0; e < deg; e++) {
    int sc = srcs[wid * deg + e];
    if (e + 1 < deg) {
      int nsc = srcs[wid * deg + e + 1];
      __builtin_prefetch(K + (size_t)nsc * DIN, 0, 0);
      __builtin_prefetch(V + (size_t)nsc * DIN, 0, 0);
    }
    v8h kv = *(const v8h*)(K + (size_t)sc * DIN + lane * 8);
    v8h vv = *(const v8h*)(V + (size_t)sc * DIN + lane * 8);
    float p = 0.f;
    #pragma unroll
    for (int i = 0; i < 8; i++) p += qr[i] * (float)kv[i];
    // per-head (16-lane) butterfly reduction
    p += __shfl_xor(p, 1, 32); p += __shfl_xor(p, 2, 32);
    p += __shfl_xor(p, 4, 32); p += __shfl_xor(p, 8, 32);
    float logit = p * scale;
    float nm   = fmaxf(mrun, logit);
    float corr = __expf(mrun - nm);
    float w    = __expf(logit - nm);
    ssum = ssum * corr + w;
    #pragma unroll
    for (int i = 0; i < 8; i++) acc[i] = acc[i] * corr + w * (float)vv[i];
    mrun = nm;
  }

  const float* sp = S + (size_t)wid * DIN + lane * 8;
  float* op = out + (size_t)wid * DIN + lane * 8;
  float inv = 1.f / ssum;
  #pragma unroll
  for (int i = 0; i < 8; i++) op[i] = acc[i] * inv + sp[i];
}

// ---------------------------------------------------------------- batchnorm

__global__ void zero512_kernel(float* p) {
  int i = blockIdx.x * blockDim.x + threadIdx.x;
  if (i < 512) p[i] = 0.f;
}

__global__ void bn_reduce_kernel(const float* __restrict__ f,
                                 float* __restrict__ sums, int M) {
  int c  = threadIdx.x;                 // 256 threads = 256 columns
  int r0 = blockIdx.x * 64;
  int r1 = r0 + 64; if (r1 > M) r1 = M;
  float s = 0.f, q = 0.f;
  for (int r = r0; r < r1; r++) {
    float v = f[(size_t)r * DIN + c];
    s += v; q += v * v;
  }
  atomicAdd(&sums[c], s);
  atomicAdd(&sums[256 + c], q);
}

__global__ void bn_apply_mish_kernel(float* __restrict__ f,
                                     const float* __restrict__ sums,
                                     const float* __restrict__ g,
                                     const float* __restrict__ b,
                                     half_t* __restrict__ fh, int M) {
  int i = blockIdx.x * blockDim.x + threadIdx.x;
  if (i >= M * DIN) return;
  int c = i & 255;
  float invM = 1.f / (float)M;
  float mean = sums[c] * invM;
  float var  = sums[256 + c] * invM - mean * mean;
  float x = f[i];
  float y = (x - mean) * rsqrtf(var + 1e-5f) * g[c] + b[c];
  float sp = (y > 20.f) ? y : log1pf(__expf(y));     // softplus, stable
  float o  = y * tanhf(sp);                           // mish
  f[i] = o;
  if (fh) fh[i] = (half_t)o;
}

// ---------------------------------------------------------------- launch

static constexpr size_t AL(size_t x) { return (x + 255) & ~(size_t)255; }

extern "C" void kernel_launch(void* const* d_in, const int* in_sizes, int n_in,
                              void* d_out, int out_size, void* d_ws, size_t ws_size,
                              hipStream_t stream) {
  const float* x    = (const float*)d_in[0];
  const int*   ei0s = (const int*)d_in[1];              // src = first N1*DEG
  const int*   ei1s = (const int*)d_in[2];              // src = first N2*DEG
  const float* Wq0 = (const float*)d_in[3];  const float* bq0 = (const float*)d_in[4];
  const float* Wk0 = (const float*)d_in[5];  const float* bk0 = (const float*)d_in[6];
  const float* Wv0 = (const float*)d_in[7];  const float* bv0 = (const float*)d_in[8];
  const float* Ws0 = (const float*)d_in[9];  const float* bs0 = (const float*)d_in[10];
  const float* g0  = (const float*)d_in[11]; const float* be0 = (const float*)d_in[12];
  const float* Wq1 = (const float*)d_in[13]; const float* bq1 = (const float*)d_in[14];
  const float* Wk1 = (const float*)d_in[15]; const float* bk1 = (const float*)d_in[16];
  const float* Wv1 = (const float*)d_in[17]; const float* bv1 = (const float*)d_in[18];
  const float* Ws1 = (const float*)d_in[19]; const float* bs1 = (const float*)d_in[20];
  const float* g1  = (const float*)d_in[21]; const float* be1 = (const float*)d_in[22];
  const float* Wout0 = (const float*)d_in[23]; const float* bout0 = (const float*)d_in[24];
  const float* Wout1 = (const float*)d_in[25]; const float* bout1 = (const float*)d_in[26];
  const float* Wout2 = (const float*)d_in[27]; const float* bout2 = (const float*)d_in[28];
  float* out = (float*)d_out;                     // [l0[:N2] | l1[:N2] | l2[:N2]]
  char* ws = (char*)d_ws;

  // ---- workspace layout (bump allocated, layer-1 aliases layer-0)
  constexpr size_t o_xh  = 0;
  constexpr size_t o_wt  = o_xh  + AL((size_t)N0 * DIN * 2);
  constexpr size_t o_l0  = o_wt  + AL((size_t)8 * 256 * 256 * 2);
  constexpr size_t o_ns  = o_l0  + AL((size_t)N0 * 4);
  constexpr size_t o_k0  = o_ns  + AL((size_t)N1 * 8 * 4);
  constexpr size_t o_v0  = o_k0  + AL((size_t)N0 * DIN * 2);
  constexpr size_t o_q0  = o_v0  + AL((size_t)N0 * DIN * 2);
  constexpr size_t o_s0  = o_q0  + AL((size_t)N1 * DIN * 4);
  constexpr size_t o_f1  = o_s0  + AL((size_t)N1 * DIN * 4);
  constexpr size_t o_f1h = o_f1  + AL((size_t)N1 * DIN * 4);
  constexpr size_t o_l1  = o_f1h + AL((size_t)N1 * DIN * 2);
  constexpr size_t o_sum = o_l1  + AL((size_t)N1 * 4);

  half_t* xh   = (half_t*)(ws + o_xh);
  half_t* wt   = (half_t*)(ws + o_wt);      // 8 matrices of 256*256 halves
  float*  l0   = (float*)(ws + o_l0);
  int*    ns0  = (int*)(ws + o_ns);
  half_t* K0h  = (half_t*)(ws + o_k0);
  half_t* V0h  = (half_t*)(ws + o_v0);
  float*  q0f  = (float*)(ws + o_q0);
  float*  s0f  = (float*)(ws + o_s0);
  float*  f1   = (float*)(ws + o_f1);
  half_t* f1h  = (half_t*)(ws + o_f1h);
  float*  l1   = (float*)(ws + o_l1);
  float*  sums = (float*)(ws + o_sum);
  // layer-1 aliases (layer-0 buffers dead by then)
  half_t* K1h = K0h;  half_t* V1h = V0h;
  float*  q1f = q0f;  float*  s1f = s0f;
  float*  f2  = (float*)(ws + o_xh);        // 20.5 MB inside 51.2 MB xh region

  const int TPB = 256;
  auto gemmGrid = [](int M) { return ((M >> 5) * 4 + 7) / 8; }; // 8 waves/block
  auto waveGrid = [](int M) { return (M + 7) / 8; };

  // 1) x -> f16
  f32_to_f16_kernel<<<(N0 * DIN + TPB - 1) / TPB, TPB, 0, stream>>>(x, xh, N0 * DIN);

  // 2) transpose+convert the 8 projection weights
  const float* Wsrc[8] = {Wq0, Wk0, Wv0, Ws0, Wq1, Wk1, Wv1, Ws1};
  for (int i = 0; i < 8; i++)
    wtrans_kernel<<<256, TPB, 0, stream>>>(Wsrc[i], wt + (size_t)i * 256 * 256);
  half_t* wtQ0 = wt + 0 * 65536; half_t* wtK0 = wt + 1 * 65536;
  half_t* wtV0 = wt + 2 * 65536; half_t* wtS0 = wt + 3 * 65536;
  half_t* wtQ1 = wt + 4 * 65536; half_t* wtK1 = wt + 5 * 65536;
  half_t* wtV1 = wt + 6 * 65536; half_t* wtS1 = wt + 7 * 65536;

  // 3) l0 = x @ Wout0 + b  (also writes d_out[0:N2])
  gemv256_kernel<<<waveGrid(N0), TPB, 0, stream>>>(x, Wout0, bout0, l0, out, N0, N2);

  // 4) label-aware pool, layer 0 (ratio 0.5 -> keep 8 of 16)
  pool0_kernel<<<(N1 + TPB - 1) / TPB, TPB, 0, stream>>>(l0, ei0s, ns0);

  // 5) layer-0 projections (WMMA)
  gemm256_wmma_kernel<<<gemmGrid(N0), TPB, 0, stream>>>(xh, wtK0, bk0, nullptr, K0h, N0);
  gemm256_wmma_kernel<<<gemmGrid(N0), TPB, 0, stream>>>(xh, wtV0, bv0, nullptr, V0h, N0);
  gemm256_wmma_kernel<<<gemmGrid(N1), TPB, 0, stream>>>(xh, wtQ0, bq0, q0f, nullptr, N1);
  gemm256_wmma_kernel<<<gemmGrid(N1), TPB, 0, stream>>>(xh, wtS0, bs0, s0f, nullptr, N1);

  // 6) attention layer 0 (deg 8) -> f1 (pre-BN)
  attn_kernel<<<waveGrid(N1), TPB, 0, stream>>>(q0f, K0h, V0h, s0f, ns0, 8, f1, N1);

  // 7) BN + mish -> f1 (in place) and f1h
  zero512_kernel<<<2, TPB, 0, stream>>>(sums);
  bn_reduce_kernel<<<(N1 + 63) / 64, TPB, 0, stream>>>(f1, sums, N1);
  bn_apply_mish_kernel<<<(N1 * DIN + TPB - 1) / TPB, TPB, 0, stream>>>(f1, sums, g0, be0, f1h, N1);

  // 8) l1 = f1 @ Wout1 + b  (also writes d_out[N2:2*N2])
  gemv256_kernel<<<waveGrid(N1), TPB, 0, stream>>>(f1, Wout1, bout1, l1, out + N2, N1, N2);

  // 9) layer-1 projections (pool ratio 1.0 is a permutation -> skip pooling)
  gemm256_wmma_kernel<<<gemmGrid(N1), TPB, 0, stream>>>(f1h, wtK1, bk1, nullptr, K1h, N1);
  gemm256_wmma_kernel<<<gemmGrid(N1), TPB, 0, stream>>>(f1h, wtV1, bv1, nullptr, V1h, N1);
  gemm256_wmma_kernel<<<gemmGrid(N2), TPB, 0, stream>>>(f1h, wtQ1, bq1, q1f, nullptr, N2);
  gemm256_wmma_kernel<<<gemmGrid(N2), TPB, 0, stream>>>(f1h, wtS1, bs1, s1f, nullptr, N2);

  // 10) attention layer 1 (deg 16, raw edge srcs) -> f2
  attn_kernel<<<waveGrid(N2), TPB, 0, stream>>>(q1f, K1h, V1h, s1f, ei1s, 16, f2, N2);

  // 11) BN + mish -> f2 (in place)
  zero512_kernel<<<2, TPB, 0, stream>>>(sums);
  bn_reduce_kernel<<<(N2 + 63) / 64, TPB, 0, stream>>>(f2, sums, N2);
  bn_apply_mish_kernel<<<(N2 * DIN + TPB - 1) / TPB, TPB, 0, stream>>>(f2, sums, g1, be1, nullptr, N2);

  // 12) l2 = f2 @ Wout2 + b -> d_out[2*N2:3*N2]
  gemv256_kernel<<<waveGrid(N2), TPB, 0, stream>>>(f2, Wout2, bout2, nullptr, out + 2 * N2, N2, N2);
}
